// LSTMWrapper_3410204033065
// MI455X (gfx1250) — compile-verified, matrix operand order checked by reading, not run on hardware
//
#include <hip/hip_runtime.h>
#include <hip/hip_bf16.h>
#include <stdint.h>

// ---------------------------------------------------------------------------
// Bidirectional 2-layer LSTM (packed-sequence semantics) for gfx1250.
//   Phase A (throughput): xg = x @ Wih^T for all (b,t) as one big bf16 WMMA
//                         GEMM; A-tile staged in LDS via async global->LDS,
//                         2Mx2N register blocking per wave.
//   Phase B (latency):    per-timestep h @ Whh^T (K=512 only) + gate math.
// All matrix math on V_WMMA_F32_16X16X32_BF16 (bf16 in, fp32 accumulate).
// ---------------------------------------------------------------------------

#define Bsz 32
#define Ssz 1024
#define Dsz 512
#define Hsz 512
#define BS  (Bsz * Ssz)
#define G4  (4 * Hsz)
#define KC  512            // K-chunk staged in LDS per iteration
#define APAD 8             // row padding (bf16 elems) -> fragment reads hit all 64 banks

#if defined(__has_builtin)
#  if __has_builtin(__builtin_amdgcn_global_load_async_to_lds_b128)
#    define USE_ASYNC_LDS 1
#  endif
#endif
#ifndef USE_ASYNC_LDS
#  define USE_ASYNC_LDS 0
#endif

typedef __attribute__((ext_vector_type(16))) __bf16 v16bf;
typedef __attribute__((ext_vector_type(8)))  float  v8f;

#if USE_ASYNC_LDS
// Builtin signature (from clang diagnostic): pointers to 16-byte int vectors,
// global (AS1) source and shared (AS3) destination, plus imm offset + cpol.
typedef __attribute__((__vector_size__(4 * sizeof(int)))) int vi4;
typedef __attribute__((address_space(1))) vi4 g_vi4;
typedef __attribute__((address_space(3))) vi4 l_vi4;
#endif

// Load one 16x32 bf16 operand fragment row for this lane.
// Lane layout (ISA 7.12.2, 16-bit A 16x32): lanes 0-15 hold K={0..7,16..23},
// lanes 16-31 hold K={8..15,24..31}; rowp points at (row, k0) in row-major bf16.
static __device__ __forceinline__ v16bf ldfrag(const unsigned short* rowp, int khalf) {
    const uint4* q0 = reinterpret_cast<const uint4*>(rowp + khalf * 8);
    const uint4* q1 = reinterpret_cast<const uint4*>(rowp + 16 + khalf * 8);
    uint4 u0 = *q0;
    uint4 u1 = *q1;
    union { uint4 u; __bf16 h[8]; } c0, c1;
    c0.u = u0; c1.u = u1;
    v16bf v;
#pragma unroll
    for (int i = 0; i < 8; ++i) {
        v[i]     = c0.h[i];
        v[8 + i] = c1.h[i];
    }
    return v;
}

static __device__ __forceinline__ unsigned short f2bf(float f) {
    union { __bf16 b; unsigned short u; } cv;
    cv.b = (__bf16)f;
    return cv.u;
}

static __device__ __forceinline__ float bf2f(unsigned short u) {
    union { unsigned short u; __bf16 b; } cv;
    cv.u = u;
    return (float)cv.b;
}

__global__ void f32_to_bf16_kernel(const float* __restrict__ in,
                                   unsigned short* __restrict__ out, int n) {
    int i = blockIdx.x * blockDim.x + threadIdx.x;
    if (i < n) out[i] = f2bf(in[i]);
}

// ---------------------------------------------------------------------------
// Phase A: xg[dir][row][4H] = x[row] @ Wih_dir^T   for all rows in [0, B*S).
// grid = (BS/32 M-blocks, 8 N-blocks * 2 dirs), block = 256 threads (8 waves).
// Block computes 32(M) x 256(N); wave w owns a 32-wide N strip with 2Mx2N
// 16x16 subtiles. The shared 32xKC A-slab lives in LDS.
// ---------------------------------------------------------------------------
__global__ __launch_bounds__(256) void xg_gemm_kernel(
    const unsigned short* __restrict__ xin, int Din,
    const unsigned short* __restrict__ WihF, const unsigned short* __restrict__ WihB,
    unsigned short* __restrict__ xg)
{
    const int wv    = threadIdx.x >> 5;       // 0..7
    const int lane  = threadIdx.x & 31;
    const int khalf = lane >> 4;
    const int mr    = lane & 15;
    const int dir   = blockIdx.y >> 3;
    const int nblk  = blockIdx.y & 7;
    const int ng0   = nblk * 256 + wv * 32 + mr;   // output feature, subtile 0
    const int ng1   = ng0 + 16;                    // subtile 1
    const int row0  = blockIdx.x * 32;             // flat (b,t) row

    const unsigned short* W     = dir ? WihB : WihF;
    const unsigned short* wrow0 = W + (size_t)ng0 * Din;
    const unsigned short* wrow1 = W + (size_t)ng1 * Din;

    __shared__ __align__(16) unsigned short ash[32][KC + APAD];

    // cooperative staging assignment: thread -> (row, 128-byte chunk)
    const int srow   = threadIdx.x >> 3;     // 0..31
    const int schunk = threadIdx.x & 7;      // 0..7  (64 bf16 elems each)

    v8f acc00 = {}, acc01 = {}, acc10 = {}, acc11 = {};

    for (int kc = 0; kc < Din; kc += KC) {
        // ---- stage A slab: x[row0..row0+31][kc..kc+KC) -> LDS ----
        {
            const unsigned short* gsrc = xin + (size_t)(row0 + srow) * Din + kc + schunk * 64;
            unsigned short*       ldst = &ash[srow][schunk * 64];
#if USE_ASYNC_LDS
#pragma unroll
            for (int i = 0; i < 8; ++i) {
                __builtin_amdgcn_global_load_async_to_lds_b128(
                    (g_vi4*)(gsrc + i * 8),
                    (l_vi4*)(ldst + i * 8),
                    0, 0);
            }
#  if __has_builtin(__builtin_amdgcn_s_wait_asynccnt)
            __builtin_amdgcn_s_wait_asynccnt(0);
#  else
            asm volatile("s_wait_asynccnt 0" ::: "memory");
#  endif
#else
#pragma unroll
            for (int i = 0; i < 8; ++i) {
                *(uint4*)(ldst + i * 8) = *(const uint4*)(gsrc + i * 8);
            }
#endif
        }
        __syncthreads();

        // ---- compute over this K chunk ----
        for (int k = 0; k < KC; k += 32) {
            v16bf a0 = ldfrag(&ash[mr][k],      khalf);   // ds_load_b128 x2
            v16bf a1 = ldfrag(&ash[mr + 16][k], khalf);
            v16bf b0 = ldfrag(wrow0 + kc + k,   khalf);   // global_load_b128 x2
            v16bf b1 = ldfrag(wrow1 + kc + k,   khalf);
            acc00 = __builtin_amdgcn_wmma_f32_16x16x32_bf16(false, a0, false, b0,
                                                            (short)0, acc00, false, false);
            acc01 = __builtin_amdgcn_wmma_f32_16x16x32_bf16(false, a0, false, b1,
                                                            (short)0, acc01, false, false);
            acc10 = __builtin_amdgcn_wmma_f32_16x16x32_bf16(false, a1, false, b0,
                                                            (short)0, acc10, false, false);
            acc11 = __builtin_amdgcn_wmma_f32_16x16x32_bf16(false, a1, false, b1,
                                                            (short)0, acc11, false, false);
        }
        __syncthreads();   // protect LDS before next stage overwrites it
    }

    unsigned short* ob = xg + (size_t)dir * BS * G4;
#pragma unroll
    for (int i = 0; i < 8; ++i) {
        int m0 = row0 + i + 8 * khalf;      // C/D layout: VGPR i -> M = i+8*khalf
        ob[(size_t)m0 * G4 + ng0]        = f2bf(acc00[i]);
        ob[(size_t)m0 * G4 + ng1]        = f2bf(acc01[i]);
        ob[(size_t)(m0 + 16) * G4 + ng0] = f2bf(acc10[i]);
        ob[(size_t)(m0 + 16) * G4 + ng1] = f2bf(acc11[i]);
    }
}

// ---------------------------------------------------------------------------
// Phase B: one recurrent timestep for one layer, both directions.
// grid = (512/16 = 32 N-tiles, 2 directions), block = 128 threads (4 waves).
// Wave w computes gate w's 32x16 tile of  h @ Whh^T;  the precomputed
// xg (input projection) and bias are folded in during the gate phase.
// ---------------------------------------------------------------------------
__global__ __launch_bounds__(128) void lstm_step_kernel(
    const unsigned short* __restrict__ xg,
    const unsigned short* __restrict__ WhhF, const unsigned short* __restrict__ WhhB,
    const float* __restrict__ biasF, const float* __restrict__ biasB,
    const int* __restrict__ lens,
    float* __restrict__ hstate, float* __restrict__ cstate,
    unsigned short* __restrict__ hbf16,
    unsigned short* __restrict__ outBf, float* __restrict__ outF, int t)
{
    const int dir   = blockIdx.y;
    const int nbase = blockIdx.x * 16;
    const int wv    = threadIdx.x >> 5;   // wave id == gate id (i,f,g,o)
    const int lane  = threadIdx.x & 31;
    const int khalf = lane >> 4;
    const int mr    = lane & 15;

    const unsigned short* Whh = dir ? WhhB : WhhF;
    const float*         bias = dir ? biasB : biasF;
    float* hs          = hstate + (size_t)dir * Bsz * Hsz;
    float* cs          = cstate + (size_t)dir * Bsz * Hsz;
    unsigned short* hb = hbf16  + (size_t)dir * Bsz * Hsz;
    const unsigned short* xgd = xg + (size_t)dir * BS * G4;

    __shared__ float gsh[4][Bsz][16];

    // B-matrix: element (k,n) = Whh[gate*H + nbase + n][k]; per lane n = mr.
    const unsigned short* whh_row = Whh + (size_t)(wv * Hsz + nbase + mr) * Hsz;
    const unsigned short* hrow0 = hb + (size_t)mr * Hsz;
    const unsigned short* hrow1 = hb + (size_t)(mr + 16) * Hsz;

    v8f acc0 = {};
    v8f acc1 = {};
#pragma unroll 4
    for (int k = 0; k < Hsz; k += 32) {
        v16bf a0 = ldfrag(hrow0 + k, khalf);
        v16bf a1 = ldfrag(hrow1 + k, khalf);
        v16bf bm = ldfrag(whh_row + k, khalf);
        acc0 = __builtin_amdgcn_wmma_f32_16x16x32_bf16(false, a0, false, bm,
                                                       (short)0, acc0, false, false);
        acc1 = __builtin_amdgcn_wmma_f32_16x16x32_bf16(false, a1, false, bm,
                                                       (short)0, acc1, false, false);
    }

    // C/D layout (ISA 7.12.2): VGPR i -> (M = i + 8*khalf, N = mr)
#pragma unroll
    for (int i = 0; i < 8; ++i) {
        gsh[wv][i + 8 * khalf][mr]      = acc0[i];
        gsh[wv][16 + i + 8 * khalf][mr] = acc1[i];
    }
    __syncthreads();

    // Elementwise gate phase: 32 rows x 16 cols = 512 outputs, 4 per thread.
#pragma unroll
    for (int e = 0; e < 4; ++e) {
        int lin = threadIdx.x + 128 * e;
        int m   = lin >> 4;          // batch row
        int n   = lin & 15;
        int col = nbase + n;         // 0..511 within this direction's H

        int  lm    = lens[m];
        bool valid = (t < lm);
        // x position this step consumed (bwd runs the sequence reversed).
        int posx = t;
        if (dir) { posx = lm - 1 - t; if (posx < 0) posx = 0; }
        const unsigned short* xr = xgd + ((size_t)m * Ssz + posx) * G4;

        float gi = gsh[0][m][n] + bias[col]           + bf2f(xr[col]);
        float gf = gsh[1][m][n] + bias[Hsz + col]     + bf2f(xr[Hsz + col]);
        float gg = gsh[2][m][n] + bias[2 * Hsz + col] + bf2f(xr[2 * Hsz + col]);
        float go = gsh[3][m][n] + bias[3 * Hsz + col] + bf2f(xr[3 * Hsz + col]);
        float si = 1.0f / (1.0f + __expf(-gi));
        float sf = 1.0f / (1.0f + __expf(-gf));
        float so = 1.0f / (1.0f + __expf(-go));
        float tg = tanhf(gg);

        size_t sidx = (size_t)m * Hsz + col;
        float c_old = cs[sidx];
        float h_old = hs[sidx];
        float c_new = sf * c_old + si * tg;
        float h_new = so * tanhf(c_new);

        float cw = valid ? c_new : c_old;
        float hw = valid ? h_new : h_old;
        cs[sidx] = cw;
        hs[sidx] = hw;
        hb[sidx] = f2bf(hw);

        // Output position: fwd writes at t; bwd un-reverses to len-1-t.
        // Masked rows (t >= len) write zeros at position t, so every (b,pos)
        // slot is written exactly once per layer/direction.
        int pos = dir ? (valid ? lm - 1 - t : t) : t;
        float oval = valid ? h_new : 0.0f;
        size_t oidx = ((size_t)m * Ssz + pos) * (2 * Hsz) + (size_t)dir * Hsz + col;
        if (outF) {
            outF[oidx] = oval;
        } else {
            outBf[oidx] = f2bf(oval);
        }
    }
}

// ---------------------------------------------------------------------------
// Host side
// ---------------------------------------------------------------------------
extern "C" void kernel_launch(void* const* d_in, const int* in_sizes, int n_in,
                              void* d_out, int out_size, void* d_ws, size_t ws_size,
                              hipStream_t stream) {
    const int*   lens   = (const int*)d_in[1];
    // dict order: input, input_lengths, Wih0f, Whh0f, b0f, Wih0b, Whh0b, b0b,
    //             Wih1f, Whh1f, b1f, Wih1b, Whh1b, b1b
    const float* bias0f = (const float*)d_in[4];
    const float* bias0b = (const float*)d_in[7];
    const float* bias1f = (const float*)d_in[10];
    const float* bias1b = (const float*)d_in[13];

    char* ws = (char*)d_ws;
    // Workspace layout (bytes; every block size is a multiple of 256):
    const size_t oX0    = 0;                                        // bf16 [B][S][512]
    const size_t oX1    = oX0 + (size_t)BS * Dsz * 2;               // bf16 [B][S][1024]
    const size_t oWih0f = oX1 + (size_t)BS * 2 * Hsz * 2;
    const size_t oWhh0f = oWih0f + (size_t)G4 * Dsz * 2;
    const size_t oWih0b = oWhh0f + (size_t)G4 * Hsz * 2;
    const size_t oWhh0b = oWih0b + (size_t)G4 * Dsz * 2;
    const size_t oWih1f = oWhh0b + (size_t)G4 * Hsz * 2;
    const size_t oWhh1f = oWih1f + (size_t)G4 * (2 * Hsz) * 2;
    const size_t oWih1b = oWhh1f + (size_t)G4 * Hsz * 2;
    const size_t oWhh1b = oWih1b + (size_t)G4 * (2 * Hsz) * 2;
    const size_t oHs    = oWhh1b + (size_t)G4 * Hsz * 2;            // h fp32 [2L][2d][B][H]
    const size_t oCs    = oHs + (size_t)2 * 2 * Bsz * Hsz * 4;      // c fp32
    const size_t oHb    = oCs + (size_t)2 * 2 * Bsz * Hsz * 4;      // h bf16 mirror
    const size_t oXg    = oHb + (size_t)2 * 2 * Bsz * Hsz * 2;      // bf16 [2dir][B*S][4H]

    unsigned short* x0b = (unsigned short*)(ws + oX0);
    unsigned short* x1b = (unsigned short*)(ws + oX1);
    float*          hS  = (float*)(ws + oHs);
    float*          cS  = (float*)(ws + oCs);
    unsigned short* hB  = (unsigned short*)(ws + oHb);
    unsigned short* xG  = (unsigned short*)(ws + oXg);

    auto cvt = [&](const void* src, size_t dstOff, size_t n) {
        f32_to_bf16_kernel<<<(unsigned)((n + 255) / 256), 256, 0, stream>>>(
            (const float*)src, (unsigned short*)(ws + dstOff), (int)n);
    };
    cvt(d_in[0],  oX0,    (size_t)BS * Dsz);
    cvt(d_in[2],  oWih0f, (size_t)G4 * Dsz);
    cvt(d_in[3],  oWhh0f, (size_t)G4 * Hsz);
    cvt(d_in[5],  oWih0b, (size_t)G4 * Dsz);
    cvt(d_in[6],  oWhh0b, (size_t)G4 * Hsz);
    cvt(d_in[8],  oWih1f, (size_t)G4 * (2 * Hsz));
    cvt(d_in[9],  oWhh1f, (size_t)G4 * Hsz);
    cvt(d_in[11], oWih1b, (size_t)G4 * (2 * Hsz));
    cvt(d_in[12], oWhh1b, (size_t)G4 * Hsz);

    (void)hipMemsetAsync(hS, 0, (size_t)2 * 2 * Bsz * Hsz * 4, stream);
    (void)hipMemsetAsync(cS, 0, (size_t)2 * 2 * Bsz * Hsz * 4, stream);
    (void)hipMemsetAsync(hB, 0, (size_t)2 * 2 * Bsz * Hsz * 2, stream);

    const dim3 gGrid(BS / 32, 16, 1);   // 1024 M-blocks x (8 N-blocks * 2 dirs)
    const dim3 gBlk(256, 1, 1);
    const dim3 sGrid(Hsz / 16, 2, 1);   // 32 N-tiles x 2 directions
    const dim3 sBlk(128, 1, 1);

    for (int layer = 0; layer < 2; ++layer) {
        const unsigned short* xin  = layer ? x1b : x0b;
        const int             Din  = layer ? (2 * Hsz) : Dsz;
        const unsigned short* WihF = (const unsigned short*)(ws + (layer ? oWih1f : oWih0f));
        const unsigned short* WihB = (const unsigned short*)(ws + (layer ? oWih1b : oWih0b));
        const unsigned short* WhhF = (const unsigned short*)(ws + (layer ? oWhh1f : oWhh0f));
        const unsigned short* WhhB = (const unsigned short*)(ws + (layer ? oWhh1b : oWhh0b));
        const float* bF = layer ? bias1f : bias0f;
        const float* bB = layer ? bias1b : bias0b;
        float*          hs = hS + (size_t)layer * 2 * Bsz * Hsz;
        float*          cs = cS + (size_t)layer * 2 * Bsz * Hsz;
        unsigned short* hb = hB + (size_t)layer * 2 * Bsz * Hsz;
        unsigned short* outBf = layer ? nullptr : x1b;
        float*          outF  = layer ? (float*)d_out : nullptr;

        // Phase A: full-occupancy input-projection GEMM for this layer.
        xg_gemm_kernel<<<gGrid, gBlk, 0, stream>>>(xin, Din, WihF, WihB, xG);

        // Phase B: latency-bound recurrence, K=512 per step only.
        for (int t = 0; t < Ssz; ++t) {
            lstm_step_kernel<<<sGrid, sBlk, 0, stream>>>(
                xG, WhhF, WhhB, bF, bB, lens,
                hs, cs, hb, outBf, outF, t);
        }
    }
}
